// ModernTransformerLayer_75385265979997
// MI455X (gfx1250) — matrix operand order repaired and checked by reading, compiled, not run on hardware
//
#include <hip/hip_runtime.h>
#include <hip/hip_bf16.h>

// ---------------------------------------------------------------------------
// ModernTransformerLayer on gfx1250 (MI455X), wave32 + WMMA f16->f32 + TDM.
// Shapes: B=2, L=2048, D=1024, H=16, Dh=64, HID=4096.
// ---------------------------------------------------------------------------

#define BB   2
#define LL   2048
#define DD   1024
#define NH   16
#define DHD  64
#define HID  4096
#define ROWS (BB * LL)          // 4096 token rows

#ifndef TDM_SIX_ARG
#define TDM_SIX_ARG 1           // this toolchain: 6-arg tensor_load_to_lds
#endif

typedef __attribute__((ext_vector_type(16))) _Float16 v16h;
typedef __attribute__((ext_vector_type(8)))  _Float16 h8;
typedef __attribute__((ext_vector_type(4)))  _Float16 h4;
typedef __attribute__((ext_vector_type(8)))  float    v8f;
typedef __attribute__((ext_vector_type(4)))  unsigned u32x4;
typedef __attribute__((ext_vector_type(8)))  int      i32x8;
typedef __attribute__((ext_vector_type(4)))  int      i32x4;

// ---------------------------------------------------------------------------
// WMMA helper:  D(16x16 f32) = A(16x32 f16) * B(32x16 f16) + C
// ---------------------------------------------------------------------------
__device__ __forceinline__ v8f wmma16(v16h a, v16h b, v8f c) {
  return __builtin_amdgcn_wmma_f32_16x16x32_f16(
      false, a, false, b, (short)0, c, false, false);
}

// A-matrix fragment (16x32 f16), row-major source: elem(m,k) = base[m*stride+k]
__device__ __forceinline__ v16h ld_afrag(const _Float16* base, int stride, int lane) {
  int m  = lane & 15;
  int ko = (lane < 16) ? 0 : 16;
  const _Float16* p = base + (size_t)m * stride;
  h8 lo = *(const h8*)(p + ko);
  h8 hi = *(const h8*)(p + ko + 8);
  v16h a;
#pragma unroll
  for (int i = 0; i < 8; ++i) { a[i] = lo[i]; a[i + 8] = hi[i]; }
  return a;
}

// B-matrix fragment (32x16 f16) from K-contiguous storage: elem(k,n)=base[n*stride+k]
__device__ __forceinline__ v16h ld_bfrag_kcont(const _Float16* base, int stride, int lane) {
  int n  = lane & 15;
  int k0 = (lane < 16) ? 0 : 16;
  const _Float16* p = base + (size_t)n * stride + k0;
  h8 lo = *(const h8*)p;
  h8 hi = *(const h8*)(p + 8);
  v16h b;
#pragma unroll
  for (int i = 0; i < 8; ++i) { b[i] = lo[i]; b[i + 8] = hi[i]; }
  return b;
}

// B-matrix fragment (32x16 f16) from N-contiguous (row-major KxN) storage
__device__ __forceinline__ v16h ld_bfrag_ncont(const _Float16* base, int stride, int lane) {
  int n  = lane & 15;
  int k0 = (lane < 16) ? 0 : 16;
  v16h b;
#pragma unroll
  for (int i = 0; i < 16; ++i) b[i] = base[(size_t)(k0 + i) * stride + n];
  return b;
}

// ---------------------------------------------------------------------------
// Tensor Data Mover: 2-D tile (tile_x elems wide, tile_y rows) of fp16 from
// global (row stride = ld_elems) into LDS at byte offset lds_off, with LDS
// row padding (pad_interval_code/pad_amount_code per D# group1 encoding).
// ---------------------------------------------------------------------------
__device__ __forceinline__ void tdm_load_2d_f16(unsigned lds_off, const void* gptr,
                                                unsigned ld_elems, unsigned rows_total,
                                                unsigned tile_x, unsigned tile_y,
                                                unsigned pad_int_code, unsigned pad_amt_code) {
  unsigned long long ga = (unsigned long long)(size_t)gptr;
  u32x4 g0;
  g0[0] = 1u;                                             // count=1, user D#
  g0[1] = lds_off;                                        // lds_addr
  g0[2] = (unsigned)(ga & 0xFFFFFFFFu);                   // global_addr lo
  g0[3] = (unsigned)((ga >> 32) & 0x1FFFFFFu) | (2u << 30); // addr hi | type=2
  i32x8 g1;
  g1[0] = (int)((1u << 16) |                              // data_size = 2B
                (1u << 20) |                              // pad_enable
                (pad_int_code << 22) | (pad_amt_code << 25));
  g1[1] = (int)((ld_elems & 0xFFFFu) << 16);              // tensor_dim0 lo16
  g1[2] = (int)(((ld_elems >> 16) & 0xFFFFu) |            // tensor_dim0 hi16
                ((rows_total & 0xFFFFu) << 16));          // tensor_dim1 lo16
  g1[3] = (int)(((rows_total >> 16) & 0xFFFFu) |          // tensor_dim1 hi16
                ((tile_x & 0xFFFFu) << 16));              // tile_dim0
  g1[4] = (int)(tile_y & 0xFFFFu);                        // tile_dim1 (tile_dim2=0)
  g1[5] = (int)ld_elems;                                  // tensor_dim0_stride lo32
  g1[6] = 0;
  g1[7] = 0;
  i32x4 z = {0, 0, 0, 0};
#if TDM_SIX_ARG
  i32x8 z8 = {0, 0, 0, 0, 0, 0, 0, 0};
  __builtin_amdgcn_tensor_load_to_lds(g0, g1, z, z, z8, 0);
#else
  __builtin_amdgcn_tensor_load_to_lds(g0, g1, z, z, 0);
#endif
}

// ---------------------------------------------------------------------------
// fp32 -> fp16 conversion
// ---------------------------------------------------------------------------
__global__ void cvt_f32_f16_kernel(const float* __restrict__ src,
                                   _Float16* __restrict__ dst, int n4) {
  int i = blockIdx.x * blockDim.x + threadIdx.x;
  int stride = gridDim.x * blockDim.x;
  for (; i < n4; i += stride) {
    float4 v = ((const float4*)src)[i];
    h4 o;
    o[0] = (_Float16)v.x; o[1] = (_Float16)v.y;
    o[2] = (_Float16)v.z; o[3] = (_Float16)v.w;
    ((h4*)dst)[i] = o;
  }
}

// ---------------------------------------------------------------------------
// RMSNorm: one 256-thread block per token row
// ---------------------------------------------------------------------------
__global__ __launch_bounds__(256)
void rmsnorm_kernel(const float* __restrict__ x, const float* __restrict__ scale,
                    _Float16* __restrict__ out) {
  const int row = blockIdx.x;
  const int tid = threadIdx.x;
  float4 xv = ((const float4*)(x + (size_t)row * DD))[tid];
  float ss = xv.x * xv.x + xv.y * xv.y + xv.z * xv.z + xv.w * xv.w;
  __shared__ float red[256];
  red[tid] = ss;
  __syncthreads();
#pragma unroll
  for (int s = 128; s > 0; s >>= 1) {
    if (tid < s) red[tid] += red[tid + s];
    __syncthreads();
  }
  float inv = rsqrtf(red[0] * (1.0f / (float)DD) + 1e-8f);
  float4 sc = ((const float4*)scale)[tid];
  h4 o;
  o[0] = (_Float16)(xv.x * sc.x * inv);
  o[1] = (_Float16)(xv.y * sc.y * inv);
  o[2] = (_Float16)(xv.z * sc.z * inv);
  o[3] = (_Float16)(xv.w * sc.w * inv);
  ((h4*)(out + (size_t)row * DD))[tid] = o;
}

// ---------------------------------------------------------------------------
// Tiled WMMA GEMM with TDM-staged B tiles and double-buffered LDS.
// Block tile 64(M) x 128(N), BLK_K=32, 256 threads = 8 waves (4 M x 2 N).
// LDS layout (dynamic, base offset 0):
//   Bs[buf]  @ buf*8704           (32 x 136 halves, padded by TDM)
//   Bs2[buf] @ 17408 + buf*8704   (MODE 2 only)
//   As[buf]  @ AOFF  + buf*5120   (64 x 40 halves)
// MODE 0: QKV scatter  MODE 1: O-proj+resid  MODE 2: SwiGLU  MODE 3: FFN2+resid
// ---------------------------------------------------------------------------
template <int MODE>
__global__ __launch_bounds__(256)
void gemm_wmma_kernel(const _Float16* __restrict__ A, const _Float16* __restrict__ W,
                      const float* __restrict__ bias, const float* __restrict__ bias2,
                      const float* __restrict__ resid, float* __restrict__ outF,
                      _Float16* __restrict__ outH,
                      _Float16* __restrict__ Qh, _Float16* __restrict__ Kh,
                      _Float16* __restrict__ Vh,
                      int M, int N, int K, int ldw) {
  extern __shared__ char smem[];
  constexpr unsigned BS_BYTES = 32 * 136 * 2;   // 8704
  constexpr unsigned AS_BYTES = 64 * 40 * 2;    // 5120
  constexpr unsigned B2_BASE  = 2 * BS_BYTES;   // 17408
  constexpr unsigned A_BASE   = (MODE == 2) ? (4 * BS_BYTES) : (2 * BS_BYTES);

  const int tid  = threadIdx.x;
  const int lane = tid & 31;
  const int wave = tid >> 5;
  const int wm   = wave & 3;
  const int wn   = wave >> 2;
  const int m0   = blockIdx.y * 64;
  const int n0   = blockIdx.x * 128;

  const int ar = tid >> 2;              // A stage: row 0..63
  const int ac = (tid & 3) * 8;         // A stage: col group
  const _Float16* Abase = A + (size_t)(m0 + ar) * K + ac;

  auto BsP  = [&](int buf) { return (_Float16*)(smem + buf * BS_BYTES); };
  auto Bs2P = [&](int buf) { return (_Float16*)(smem + B2_BASE + buf * BS_BYTES); };
  auto AsP  = [&](int buf) { return (_Float16*)(smem + A_BASE + buf * AS_BYTES); };

  auto issueB = [&](int k0, int buf) {
    // B tile: 32 rows x 128 halves, LDS rows padded 256B->272B (64DW + 4DW)
    tdm_load_2d_f16(buf * BS_BYTES, W + (size_t)k0 * ldw + n0,
                    (unsigned)ldw, (unsigned)K, 128, 32, /*pad*/ 5, 3);
    if constexpr (MODE == 2)
      tdm_load_2d_f16(B2_BASE + buf * BS_BYTES, W + (size_t)k0 * ldw + n0 + HID,
                      (unsigned)ldw, (unsigned)K, 128, 32, 5, 3);
  };

  v8f acc[4] = {{}, {}, {}, {}};
  v8f acc2[4] = {{}, {}, {}, {}};

  // ---- prologue: stage tile 0 ----
  if (wave == 0) issueB(0, 0);
  {
    float4 a0 = *(const float4*)(Abase);
    *(float4*)(AsP(0) + ar * 40 + ac) = a0;
  }
  if (wave == 0) __builtin_amdgcn_s_wait_tensorcnt((short)0);
  __syncthreads();

  for (int k0 = 0; k0 < K; k0 += 32) {
    const int buf = (k0 >> 5) & 1;
    const int nxt = buf ^ 1;
    const bool more = (k0 + 32) < K;

    float4 areg;
    if (more) {
      areg = *(const float4*)(Abase + k0 + 32);
      if (wave == 0) issueB(k0 + 32, nxt);
      __builtin_prefetch(Abase + k0 + 96, 0, 0);
    }

    // ---- compute on current buffer ----
    v16h a = ld_afrag(AsP(buf) + wm * 16 * 40, 40, lane);
#pragma unroll
    for (int t = 0; t < 4; ++t) {
      v16h b = ld_bfrag_ncont(BsP(buf) + wn * 64 + t * 16, 136, lane);
      acc[t] = wmma16(a, b, acc[t]);
    }
    if constexpr (MODE == 2) {
#pragma unroll
      for (int t = 0; t < 4; ++t) {
        v16h b = ld_bfrag_ncont(Bs2P(buf) + wn * 64 + t * 16, 136, lane);
        acc2[t] = wmma16(a, b, acc2[t]);
      }
    }

    if (more) {
      *(float4*)(AsP(nxt) + ar * 40 + ac) = areg;
      if (wave == 0) __builtin_amdgcn_s_wait_tensorcnt((short)0);
    }
    __syncthreads();
  }

  // ---- epilogue ----
  const int halfg = lane >> 4;
  const int n_in  = lane & 15;
#pragma unroll
  for (int t = 0; t < 4; ++t) {
#pragma unroll
    for (int vr = 0; vr < 8; ++vr) {
      const int row = m0 + wm * 16 + halfg * 8 + vr;
      const int col = n0 + wn * 64 + t * 16 + n_in;
      const float v = acc[t][vr];
      if constexpr (MODE == 0) {
        const int sel = col >> 10;
        const int d   = col & (DD - 1);
        const int hh  = d >> 6;
        const int dh  = d & (DHD - 1);
        const int bb  = row >> 11;
        const int l   = row & (LL - 1);
        _Float16* dst = (sel == 0) ? Qh : (sel == 1) ? Kh : Vh;
        dst[(((size_t)bb * NH + hh) * LL + l) * DHD + dh] =
            (_Float16)(v + bias[col]);
      } else if constexpr (MODE == 1) {
        outF[(size_t)row * DD + col] =
            resid[(size_t)row * DD + col] + v + bias[col];
      } else if constexpr (MODE == 2) {
        const float xp = v + bias[col];
        const float g  = acc2[t][vr] + bias2[col];
        const float si = g / (1.0f + __expf(-g));
        outH[(size_t)row * HID + col] = (_Float16)(si * xp);
      } else {
        outF[(size_t)row * DD + col] =
            resid[(size_t)row * DD + col] + v + bias[col];
      }
    }
  }
}

// ---------------------------------------------------------------------------
// Flash attention, all-WMMA, TDM-staged K tiles, running ALiBi bias.
// grid = (B*H, L/64); block = 128 (4 waves x 16 q rows); 32-key steps.
// Dynamic LDS: Ks @ 0 (32x72h), Vt @ 4608 (64x40h, transposed), Ps @ 9728.
// ---------------------------------------------------------------------------
__global__ __launch_bounds__(128)
void attn_wmma_kernel(const _Float16* __restrict__ Qh,
                      const _Float16* __restrict__ Kh,
                      const _Float16* __restrict__ Vh,
                      _Float16* __restrict__ outH) {
  extern __shared__ char smem[];
  _Float16* Ks = (_Float16*)smem;                    // stride 72
  _Float16* Vt = (_Float16*)(smem + 4608);           // stride 40
  _Float16* Ps = (_Float16*)(smem + 9728);           // per-wave 16x40

  const int tid  = threadIdx.x;
  const int lane = tid & 31;
  const int wave = tid >> 5;
  const int bh   = blockIdx.x;
  const int h    = bh & (NH - 1);
  const int b    = bh >> 4;
  const int q0   = blockIdx.y * 64;
  const int qbase = q0 + wave * 16;
  const float slope = exp2f(-0.5f * (float)(h + 1));

  const size_t headoff = (size_t)bh * LL * DHD;
  const _Float16* Qp = Qh + headoff;
  const _Float16* Kp = Kh + headoff;
  const _Float16* Vp = Vh + headoff;

  v16h aq0 = ld_afrag(Qp + (size_t)qbase * DHD + 0, DHD, lane);
  v16h aq1 = ld_afrag(Qp + (size_t)qbase * DHD + 32, DHD, lane);

  v8f o[4] = {{}, {}, {}, {}};
  float mi[8], li[8];
  float bal[2][8];                 // running ALiBi bias: -slope*(i - j)
  const int halfg = lane >> 4;
  const int n_in  = lane & 15;
#pragma unroll
  for (int vr = 0; vr < 8; ++vr) {
    mi[vr] = -1e30f;
    li[vr] = 0.0f;
    const int i = qbase + halfg * 8 + vr;
    bal[0][vr] = slope * (float)((0 * 16 + n_in) - i);
    bal[1][vr] = slope * (float)((1 * 16 + n_in) - i);
  }
  const float bstep = slope * 32.0f;
  _Float16* Pw = Ps + wave * 16 * 40;

  const int nkb = q0 / 32 + 2;
  for (int kb = 0; kb < nkb; ++kb) {
    const int j0 = kb * 32;
    // --- stage: K via TDM (wave 0), V transposed by all threads ---
    if (wave == 0)
      tdm_load_2d_f16(/*lds_off=*/0, Kp + (size_t)j0 * DHD,
                      DHD, LL, 64, 32, /*pad: 32DW rows + 4DW*/ 4, 3);
    {
      int key = tid >> 2;
      int g   = (tid & 3) * 16;
      const float4* vsrc = (const float4*)(Vp + (size_t)(j0 + key) * DHD + g);
      float4 v0 = vsrc[0], v1 = vsrc[1];
      const _Float16* vh0 = (const _Float16*)&v0;
      const _Float16* vh1 = (const _Float16*)&v1;
#pragma unroll
      for (int ii = 0; ii < 8; ++ii) Vt[(g + ii) * 40 + key] = vh0[ii];
#pragma unroll
      for (int ii = 0; ii < 8; ++ii) Vt[(g + 8 + ii) * 40 + key] = vh1[ii];
    }
    if (wave == 0) __builtin_amdgcn_s_wait_tensorcnt((short)0);
    __syncthreads();

    // --- scores: two 16x16 tiles ---
    float sv[2][8];
#pragma unroll
    for (int st = 0; st < 2; ++st) {
      v16h bk0 = ld_bfrag_kcont(Ks + (st * 16) * 72 + 0, 72, lane);
      v16h bk1 = ld_bfrag_kcont(Ks + (st * 16) * 72 + 32, 72, lane);
      v8f z = {};
      v8f s = wmma16(aq0, bk0, z);
      s = wmma16(aq1, bk1, s);
#pragma unroll
      for (int vr = 0; vr < 8; ++vr) {
        const float bias = bal[st][vr];          // == -slope*(i-j)
        const float t = s[vr] * 0.125f + bias;   // 1/sqrt(64)
        sv[st][vr] = (bias <= 0.0f) ? t : -1e30f;  // causal: j<=i <=> bias<=0
        bal[st][vr] = bias + bstep;              // advance to next key block
      }
    }

    // --- online softmax over this 32-key block ---
#pragma unroll
    for (int vr = 0; vr < 8; ++vr) {
      float rm = fmaxf(sv[0][vr], sv[1][vr]);
#pragma unroll
      for (int off = 8; off > 0; off >>= 1)
        rm = fmaxf(rm, __shfl_xor(rm, off, 32));
      const float mn = fmaxf(mi[vr], rm);
      const float p0 = __expf(sv[0][vr] - mn);
      const float p1 = __expf(sv[1][vr] - mn);
      float rs = p0 + p1;
#pragma unroll
      for (int off = 8; off > 0; off >>= 1)
        rs += __shfl_xor(rs, off, 32);
      const float alpha = __expf(mi[vr] - mn);
      li[vr] = alpha * li[vr] + rs;
      mi[vr] = mn;
#pragma unroll
      for (int t = 0; t < 4; ++t) o[t][vr] *= alpha;
      const int r = halfg * 8 + vr;
      Pw[r * 40 + n_in]      = (_Float16)p0;
      Pw[r * 40 + 16 + n_in] = (_Float16)p1;
    }
    __syncthreads();

    // --- O += P(16x32) @ V(32x64) ---
    v16h pa = ld_afrag(Pw, 40, lane);
#pragma unroll
    for (int t = 0; t < 4; ++t) {
      v16h bv = ld_bfrag_kcont(Vt + (t * 16) * 40, 40, lane);
      o[t] = wmma16(pa, bv, o[t]);
    }
    __syncthreads();
  }

  // --- normalize + store ---
#pragma unroll
  for (int t = 0; t < 4; ++t) {
#pragma unroll
    for (int vr = 0; vr < 8; ++vr) {
      const int m  = qbase + halfg * 8 + vr;
      const int dh = t * 16 + n_in;
      const float val = o[t][vr] / li[vr];
      outH[((size_t)b * LL + m) * DD + h * DHD + dh] = (_Float16)val;
    }
  }
}

// ---------------------------------------------------------------------------
// Host orchestration
// ---------------------------------------------------------------------------
extern "C" void kernel_launch(void* const* d_in, const int* in_sizes, int n_in,
                              void* d_out, int out_size, void* d_ws, size_t ws_size,
                              hipStream_t stream) {
  (void)in_sizes; (void)n_in; (void)out_size; (void)ws_size;

  const float* x      = (const float*)d_in[0];
  // d_in[1] attn_mask, d_in[2] alibi_bias: recomputed in-kernel
  const float* w_qkv  = (const float*)d_in[3];
  const float* b_qkv  = (const float*)d_in[4];
  const float* w_o    = (const float*)d_in[5];
  const float* b_o    = (const float*)d_in[6];
  const float* scale1 = (const float*)d_in[7];
  const float* scale2 = (const float*)d_in[8];
  const float* w1     = (const float*)d_in[9];
  const float* b1     = (const float*)d_in[10];
  const float* w2     = (const float*)d_in[11];
  const float* b2     = (const float*)d_in[12];
  float* out = (float*)d_out;

  // -------- workspace carve --------
  char* ws = (char*)d_ws;
  size_t off = 0;
  auto alloc = [&](size_t bytes) {
    void* p = ws + off;
    off = (off + bytes + 255) & ~(size_t)255;
    return p;
  };
  _Float16* w_qkv_h = (_Float16*)alloc((size_t)DD * 3 * DD * 2);
  _Float16* w_o_h   = (_Float16*)alloc((size_t)DD * DD * 2);
  _Float16* w1_h    = (_Float16*)alloc((size_t)DD * 2 * HID * 2);
  _Float16* w2_h    = (_Float16*)alloc((size_t)HID * DD * 2);
  _Float16* xn_h    = (_Float16*)alloc((size_t)ROWS * DD * 2);
  _Float16* Qh      = (_Float16*)alloc((size_t)BB * NH * LL * DHD * 2);
  _Float16* Kh      = (_Float16*)alloc((size_t)BB * NH * LL * DHD * 2);
  _Float16* Vh      = (_Float16*)alloc((size_t)BB * NH * LL * DHD * 2);
  _Float16* attn_h  = (_Float16*)alloc((size_t)ROWS * DD * 2);
  float*    x2      = (float*)  alloc((size_t)ROWS * DD * 4);
  _Float16* x2n_h   = (_Float16*)alloc((size_t)ROWS * DD * 2);
  _Float16* ffn_h   = (_Float16*)alloc((size_t)ROWS * HID * 2);

  // -------- 1) weight fp32->fp16 --------
  auto cvt = [&](const float* s, _Float16* d, size_t n) {
    int n4 = (int)(n / 4);
    int blocks = (n4 + 255) / 256;
    if (blocks > 4096) blocks = 4096;
    cvt_f32_f16_kernel<<<blocks, 256, 0, stream>>>(s, d, n4);
  };
  cvt(w_qkv, w_qkv_h, (size_t)DD * 3 * DD);
  cvt(w_o,   w_o_h,   (size_t)DD * DD);
  cvt(w1,    w1_h,    (size_t)DD * 2 * HID);
  cvt(w2,    w2_h,    (size_t)HID * DD);

  // -------- 2) RMSNorm(x, scale1) --------
  rmsnorm_kernel<<<ROWS, 256, 0, stream>>>(x, scale1, xn_h);

  constexpr unsigned GEMM_SMEM  = 2 * 8704 + 2 * 5120;            // 27648
  constexpr unsigned GEMM_SMEM2 = 4 * 8704 + 2 * 5120;            // 45056
  constexpr unsigned ATTN_SMEM  = 4608 + 5120 + 4 * 16 * 40 * 2;  // 14848

  // -------- 3) QKV projection --------
  gemm_wmma_kernel<0><<<dim3(3 * DD / 128, ROWS / 64), 256, GEMM_SMEM, stream>>>(
      xn_h, w_qkv_h, b_qkv, nullptr, nullptr, nullptr, nullptr,
      Qh, Kh, Vh, ROWS, 3 * DD, DD, 3 * DD);

  // -------- 4) attention --------
  attn_wmma_kernel<<<dim3(BB * NH, LL / 64), 128, ATTN_SMEM, stream>>>(
      Qh, Kh, Vh, attn_h);

  // -------- 5) output projection + residual --------
  gemm_wmma_kernel<1><<<dim3(DD / 128, ROWS / 64), 256, GEMM_SMEM, stream>>>(
      attn_h, w_o_h, b_o, nullptr, x, x2, nullptr,
      nullptr, nullptr, nullptr, ROWS, DD, DD, DD);

  // -------- 6) RMSNorm(x2, scale2) --------
  rmsnorm_kernel<<<ROWS, 256, 0, stream>>>(x2, scale2, x2n_h);

  // -------- 7) FFN1 dual-GEMM + fused SwiGLU --------
  gemm_wmma_kernel<2><<<dim3(HID / 128, ROWS / 64), 256, GEMM_SMEM2, stream>>>(
      x2n_h, w1_h, b1, b1 + HID, nullptr, nullptr, ffn_h,
      nullptr, nullptr, nullptr, ROWS, HID, DD, 2 * HID);

  // -------- 8) FFN2 + residual -> d_out --------
  gemm_wmma_kernel<3><<<dim3(DD / 128, ROWS / 64), 256, GEMM_SMEM, stream>>>(
      ffn_h, w2_h, b2, nullptr, x2, out, nullptr,
      nullptr, nullptr, nullptr, ROWS, DD, HID, DD);
}
